// BDHParallelAttention_70308614636273
// MI455X (gfx1250) — compile-verified, matrix operand order checked by reading, not run on hardware
//
#include <hip/hip_runtime.h>

// CDNA5 (gfx1250) chunked linear attention with RoPE.
// Pass 1: RoPE(Q)->bf16 (row + transposed layouts), V->bf16 transposed (ws).
// Pass 2: per (head, D-block) workgroup scan over 32 chunks:
//         TDM tensor_load_to_lds stages tiles, bf16 WMMA + f32 accumulators.

typedef __attribute__((ext_vector_type(16))) __bf16 v16bf;
typedef __attribute__((ext_vector_type(8)))  float  v8f;
typedef __attribute__((ext_vector_type(4)))  unsigned int v4u;
typedef __attribute__((ext_vector_type(8)))  int v8i;
typedef __attribute__((ext_vector_type(4)))  int v4i;

#define NHEAD 16
#define TSEQ  4096
#define NDIM  256
#define DDIM  1024
#define CHNK  128
#define NCH   32
#define DB    64      // D-block per workgroup
#define NW    8       // waves per workgroup (wave32)

// LDS row strides (elements): data + 8-element (16B) pad per row
#define QS_S 264   // qs  [128][264] : q row-major [c][n]
#define QT_S 136   // qsT [256][136] : q^T         [n][c]
#define ST_S 264   // stb [ 64][264] : state^T     [d][n]
#define VT_S 136   // vsT [ 64][136] : v^T         [d][c]
#define SC_S 136   // sc  [128][136] : scores      [c][k]

// workspace offsets (bytes)
#define WS_QHAT   0ull
#define WS_QHATT  33554432ull   // 16*4096*256*2
#define WS_VT     67108864ull   // + 16*4096*256*2   (Vt: 1024*4096*2 = 8 MB)

// ---------------------------------------------------------------- fragments
__device__ __forceinline__ v16bf ldfrag(const __bf16* p0, int S) {
  const int lane = threadIdx.x & 31;
  const int ln = lane & 15;
  const int hi = lane >> 4;
  const __bf16* p = p0 + ln * S + hi * 8;
  union { uint4 u[2]; v16bf v; } r;
  r.u[0] = *(const uint4*)(p);
  r.u[1] = *(const uint4*)(p + 16);
  return r.v;
}

__device__ __forceinline__ v8f wmma_bf16(v16bf a, v16bf b, v8f c) {
  return __builtin_amdgcn_wmma_f32_16x16x32_bf16(false, a, false, b, (short)0, c,
                                                 false, false);
}

// ------------------------------------------------------------------- TDM
// 2D bf16 tile DMA: global (row-major, stride0 elements/row) -> LDS with a
// fixed pad inserted after every row (D# pad_enable).  ISA 8.3/8.4 encoding.
// Toolchain uses the 6-arg builtin: (v4u g0, v8i g1, v4i g2, v4i g3, v8i, cpol)
__device__ __forceinline__ void tdm_load_2d(unsigned lds_byte_off,
                                            const __bf16* gptr,
                                            unsigned tensor_d0,
                                            unsigned tensor_d1,
                                            unsigned stride0,
                                            unsigned tile_d0,
                                            unsigned tile_d1,
                                            unsigned pad_interval_code,
                                            unsigned pad_amount_code) {
  const unsigned long long ga = (unsigned long long)(uintptr_t)gptr;
  v4u g0;
  g0.x = 1u;                                    // count=1, no gather
  g0.y = lds_byte_off;                          // lds_addr [63:32]
  g0.z = (unsigned)(ga & 0xffffffffu);          // global_addr lo
  g0.w = (unsigned)((ga >> 32) & 0x01ffffffu)   // global_addr hi (56:32)
         | (2u << 30);                          // type=2 ("image")
  v8i g1;
  g1[0] = (int)((1u << 16)                      // data_size = 2 bytes
              | (1u << 20)                      // pad_enable
              | (pad_interval_code << 22)
              | (pad_amount_code << 25));
  g1[1] = (int)((tensor_d0 & 0xffffu) << 16);                 // td0[15:0]
  g1[2] = (int)(((tensor_d0 >> 16) & 0xffffu)                 // td0[31:16]
              | ((tensor_d1 & 0xffffu) << 16));               // td1[15:0]
  g1[3] = (int)(((tensor_d1 >> 16) & 0xffffu)                 // td1[31:16]
              | ((tile_d0 & 0xffffu) << 16));                 // tile_dim0
  g1[4] = (int)(tile_d1 & 0xffffu);                           // tile_dim1 (d2=0)
  g1[5] = (int)stride0;                                       // dim0_stride lo
  g1[6] = 0;                                                  // stride hi / d1s
  g1[7] = 0;
  const v4i z4 = {0, 0, 0, 0};
  const v8i z8 = {0, 0, 0, 0, 0, 0, 0, 0};
  __builtin_amdgcn_tensor_load_to_lds(g0, g1, z4, z4, z8, 0);
}

// ------------------------------------------------------------ prep kernels
__global__ __launch_bounds__(256)
void rope_prep_kernel(const float* __restrict__ Q,
                      __bf16* __restrict__ Qhat,
                      __bf16* __restrict__ QhatT) {
  const int g = blockIdx.x * 256 + threadIdx.x;   // 0 .. 131071
  const int row = g >> 1;                         // (h,t) row 0..65535
  const int half = g & 1;
  const int h = row >> 12;
  const int t = row & (TSEQ - 1);
  const float pos = (float)t;
  const float2* qp = (const float2*)(Q + (size_t)row * NDIM) + half * 64;
  __bf16* qh = Qhat + (size_t)row * NDIM;
#pragma unroll 4
  for (int i = 0; i < 64; ++i) {
    const float2 qv = qp[i];
    const int ipair = half * 64 + i;
    const int n = 2 * ipair;
    // freq = theta^(-2*ipair/N) / (2*pi), theta = 2^16, N = 256
    const float f =
        __builtin_exp2f(-(float)ipair * 0.125f) * 0.15915494309189535f;
    float ph = pos * f;
    ph = (ph - floorf(ph)) * 6.28318530717958647692f;
    float sn, cn;
    __sincosf(ph, &sn, &cn);
    const float oe = qv.x * cn - qv.y * sn;
    const float oo = qv.y * cn + qv.x * sn;
    qh[n]     = (__bf16)oe;
    qh[n + 1] = (__bf16)oo;
    QhatT[((size_t)(h * NDIM + n)) * TSEQ + t]     = (__bf16)oe;
    QhatT[((size_t)(h * NDIM + n + 1)) * TSEQ + t] = (__bf16)oo;
  }
}

__global__ __launch_bounds__(256)
void v_transpose_kernel(const float* __restrict__ V,
                        __bf16* __restrict__ Vt) {
  const int g = blockIdx.x * 256 + threadIdx.x;   // 0 .. 4194303
  const int t = g >> 10;
  const int d = g & (DDIM - 1);
  Vt[(size_t)d * TSEQ + t] = (__bf16)V[g];
}

// ------------------------------------------------------------- main kernel
__global__ __launch_bounds__(256, 1)
void bdh_linattn_kernel(const __bf16* __restrict__ Qhat,
                        const __bf16* __restrict__ QhatT,
                        const __bf16* __restrict__ Vt,
                        float* __restrict__ O) {
  __shared__ __align__(16) __bf16 qs [CHNK][QS_S];
  __shared__ __align__(16) __bf16 qsT[NDIM][QT_S];
  __shared__ __align__(16) __bf16 stb[DB  ][ST_S];
  __shared__ __align__(16) __bf16 vsT[DB  ][VT_S];
  __shared__ __align__(16) __bf16 sc [CHNK][SC_S];

  const int h     = blockIdx.x >> 4;
  const int dbase = (blockIdx.x & 15) * DB;
  const int tid   = threadIdx.x;
  const int w     = tid >> 5;
  const int lane  = tid & 31;
  const int ln    = lane & 15;
  const int hi    = lane >> 4;

  // generic->LDS: low 32 bits of a flat LDS address are the LDS byte offset
  const unsigned qs_off  = (unsigned)(uintptr_t)&qs[0][0];
  const unsigned qsT_off = (unsigned)(uintptr_t)&qsT[0][0];
  const unsigned vsT_off = (unsigned)(uintptr_t)&vsT[0][0];

  // state slice accumulators: wave w owns n-tiles {2w,2w+1} x d-tiles {0..3}
  v8f sacc[2][4];
#pragma unroll
  for (int a = 0; a < 2; ++a)
#pragma unroll
    for (int b = 0; b < 4; ++b)
      sacc[a][b] = v8f{0.f, 0.f, 0.f, 0.f, 0.f, 0.f, 0.f, 0.f};

#pragma unroll 1
  for (int ch = 0; ch < NCH; ++ch) {
    __syncthreads();  // all waves done with previous chunk's LDS

    // ---- TDM: stage q, q^T, v^T tiles for this chunk (wave 0 issues) ----
    if (w == 0) {
      // qs[128][256+pad]: rows 512B -> pad_interval 2^6*8B, pad 4 DWORDs
      tdm_load_2d(qs_off,
                  Qhat + ((size_t)(h * TSEQ + ch * CHNK)) * NDIM,
                  NDIM, TSEQ, NDIM, NDIM, CHNK, 6u, 3u);
      // qsT[256][128+pad]: rows 256B -> pad_interval 2^5*8B
      tdm_load_2d(qsT_off,
                  QhatT + (size_t)(h * NDIM) * TSEQ + ch * CHNK,
                  TSEQ, NDIM, TSEQ, CHNK, NDIM, 5u, 3u);
      // vsT[64][128+pad]
      tdm_load_2d(vsT_off,
                  Vt + (size_t)dbase * TSEQ + ch * CHNK,
                  TSEQ, DDIM, TSEQ, CHNK, DB, 5u, 3u);
    }

    // ---- overlap: snapshot state (bf16) into LDS while DMA is in flight
#pragma unroll
    for (int nt2 = 0; nt2 < 2; ++nt2) {
      const int nt = 2 * w + nt2;
#pragma unroll
      for (int dt = 0; dt < 4; ++dt) {
        __align__(16) __bf16 tmp[8];
#pragma unroll
        for (int j = 0; j < 8; ++j) tmp[j] = (__bf16)sacc[nt2][dt][j];
        *(uint4*)&stb[dt * 16 + ln][nt * 16 + hi * 8] = *(const uint4*)tmp;
      }
    }

    if (w == 0) __builtin_amdgcn_s_wait_tensorcnt(0);
    __syncthreads();  // TDM tiles + state snapshot visible to all waves

    // ---- scores = tril_{-1}(q @ q^T), wave w: rows 16w..16w+15 ----
#pragma unroll
    for (int kt = 0; kt < 8; ++kt) {
      v8f acc = v8f{0.f, 0.f, 0.f, 0.f, 0.f, 0.f, 0.f, 0.f};
#pragma unroll
      for (int nn = 0; nn < 8; ++nn) {
        const v16bf a = ldfrag(&qs[w * 16][nn * 32], QS_S);
        const v16bf b = ldfrag(&qs[kt * 16][nn * 32], QS_S);
        acc = wmma_bf16(a, b, acc);
      }
#pragma unroll
      for (int j = 0; j < 8; ++j) {
        const int crow = w * 16 + j + 8 * hi;
        const int kcol = kt * 16 + ln;
        const float val = (kcol < crow) ? acc[j] : 0.0f;  // strict lower
        sc[crow][kcol] = (__bf16)val;
      }
    }

    // ---- out_inter = q @ state ----
    v8f oacc[4];
#pragma unroll
    for (int dt = 0; dt < 4; ++dt)
      oacc[dt] = v8f{0.f, 0.f, 0.f, 0.f, 0.f, 0.f, 0.f, 0.f};
#pragma unroll
    for (int nn = 0; nn < 8; ++nn) {
      const v16bf a = ldfrag(&qs[w * 16][nn * 32], QS_S);
#pragma unroll
      for (int dt = 0; dt < 4; ++dt) {
        const v16bf b = ldfrag(&stb[dt * 16][nn * 32], ST_S);
        oacc[dt] = wmma_bf16(a, b, oacc[dt]);
      }
    }

    // ---- state += q^T @ v ----
#pragma unroll
    for (int nt2 = 0; nt2 < 2; ++nt2) {
      const int nt = 2 * w + nt2;
#pragma unroll
      for (int kk = 0; kk < 4; ++kk) {
        const v16bf a = ldfrag(&qsT[nt * 16][kk * 32], QT_S);
#pragma unroll
        for (int dt = 0; dt < 4; ++dt) {
          const v16bf b = ldfrag(&vsT[dt * 16][kk * 32], VT_S);
          sacc[nt2][dt] = wmma_bf16(a, b, sacc[nt2][dt]);
        }
      }
    }
    __syncthreads();  // scores fully in LDS

    // ---- out += scores @ v ----
#pragma unroll
    for (int kk = 0; kk < 4; ++kk) {
      const v16bf a = ldfrag(&sc[w * 16][kk * 32], SC_S);
#pragma unroll
      for (int dt = 0; dt < 4; ++dt) {
        const v16bf b = ldfrag(&vsT[dt * 16][kk * 32], VT_S);
        oacc[dt] = wmma_bf16(a, b, oacc[dt]);
      }
    }

    // ---- store out chunk slice ----
#pragma unroll
    for (int dt = 0; dt < 4; ++dt) {
#pragma unroll
      for (int j = 0; j < 8; ++j) {
        const int row = ch * CHNK + w * 16 + j + 8 * hi;
        const int col = dbase + dt * 16 + ln;
        O[(size_t)(h * TSEQ + row) * DDIM + col] = oacc[dt][j];
      }
    }
  }
}

// -------------------------------------------------------------- launcher
extern "C" void kernel_launch(void* const* d_in, const int* in_sizes, int n_in,
                              void* d_out, int out_size, void* d_ws,
                              size_t ws_size, hipStream_t stream) {
  (void)in_sizes; (void)n_in; (void)out_size; (void)ws_size;  // ws >= 72 MB
  const float* Q = (const float*)d_in[0];  // [16,4096,256] f32
  const float* V = (const float*)d_in[1];  // [4096,1024]  f32
  float* O = (float*)d_out;                // [16,4096,1024] f32

  __bf16* qhat  = (__bf16*)((char*)d_ws + WS_QHAT);
  __bf16* qhatT = (__bf16*)((char*)d_ws + WS_QHATT);
  __bf16* vt    = (__bf16*)((char*)d_ws + WS_VT);

  rope_prep_kernel<<<dim3(512), dim3(256), 0, stream>>>(Q, qhat, qhatT);
  v_transpose_kernel<<<dim3(16384), dim3(256), 0, stream>>>(V, vt);
  bdh_linattn_kernel<<<dim3(NHEAD * (DDIM / DB)), dim3(256), 0, stream>>>(
      qhat, qhatT, vt, O);
}
